// SoftAlignedDiceLoss_63196148793508
// MI455X (gfx1250) — matrix operand order repaired and checked by reading, not compiled, not run on hardware
//
#include <hip/hip_runtime.h>
#include <hip/hip_bf16.h>

typedef __attribute__((ext_vector_type(16))) __bf16 v16bf;
typedef __attribute__((ext_vector_type(8)))  float  v8f;

// Problem geometry
#define HH 512
#define WW 512
#define NIMG 64
#define RROWS 32                 // rows owned per workgroup
#define TROWS (RROWS + 4)        // staged target rows (+/-2 halo)
#define NBLK (HH / RROWS)        // 16 row-blocks per image
#define NWG  (NIMG * NBLK)       // 1024 workgroups

__global__ __launch_bounds__(256)
void dice_main(const float* __restrict__ inp,
               const float* __restrict__ tgt,
               float* __restrict__ wsf) {
    __shared__ __bf16 sT[TROWS][WW];       // target rows (bf16)
    __shared__ __bf16 sI[RROWS][WW + 8];   // input rows, col halo: sI[r][2+col]
    __shared__ float  sInter[25];
    __shared__ float  sSum[2];             // [0]=i_sum partial, [1]=t_sum partial

    const int tid = threadIdx.x;
    const int blk = blockIdx.x;            // 0..15 row block
    const int img = blockIdx.y;            // 0..63 image
    const int r0  = blk * RROWS;
    const size_t ibase = (size_t)img * HH * WW;

    float isum = 0.f, tsum = 0.f;

    // ---- Stage target rows (r0-2 .. r0+33) mod 512, convert to bf16 ----
    for (int idx = tid; idx < TROWS * (WW / 4); idx += 256) {
        int j  = idx / (WW / 4);
        int c4 = (idx - j * (WW / 4)) * 4;
        int grow = (r0 + (HH - 2) + j) & (HH - 1);      // wrap within image
        const float4 v = *(const float4*)(tgt + ibase + (size_t)grow * WW + c4);
        sT[j][c4 + 0] = (__bf16)v.x;
        sT[j][c4 + 1] = (__bf16)v.y;
        sT[j][c4 + 2] = (__bf16)v.z;
        sT[j][c4 + 3] = (__bf16)v.w;
        if (j >= 2 && j < TROWS - 2)                    // owned rows only
            tsum += (v.x + v.y) + (v.z + v.w);
    }

    // ---- Stage input rows r0..r0+31 (main body), convert to bf16 ----
    for (int idx = tid; idx < RROWS * (WW / 4); idx += 256) {
        int r  = idx / (WW / 4);
        int c4 = (idx - r * (WW / 4)) * 4;
        const float4 v = *(const float4*)(inp + ibase + (size_t)(r0 + r) * WW + c4);
        sI[r][2 + c4 + 0] = (__bf16)v.x;
        sI[r][2 + c4 + 1] = (__bf16)v.y;
        sI[r][2 + c4 + 2] = (__bf16)v.z;
        sI[r][2 + c4 + 3] = (__bf16)v.w;
        isum += (v.x + v.y) + (v.z + v.w);
    }
    // Column halo (wrap): sI[r][0,1] = cols 510,511 ; sI[r][514,515] = cols 0,1
    for (int idx = tid; idx < RROWS * 4; idx += 256) {
        int r  = idx >> 2;
        int hh = idx & 3;
        int col = (hh < 2) ? (WW - 2 + hh) : (hh - 2);
        int p   = (hh < 2) ? hh : (WW + hh);
        sI[r][p] = (__bf16)inp[ibase + (size_t)(r0 + r) * WW + col];
    }

    if (tid < 25) sInter[tid] = 0.f;
    if (tid >= 25 && tid < 27) sSum[tid - 25] = 0.f;
    __syncthreads();

    atomicAdd(&sSum[0], isum);   // ds_add_f32
    atomicAdd(&sSum[1], tsum);

    // ---- WMMA compute: wave w handles local rows 4w..4w+3 ----
    const int wv   = tid >> 5;
    const int lane = tid & 31;
    const int half = lane >> 4;          // K-group selector
    const int q    = lane & 15;
    const int idx5 = (q < 5) ? q : 4;    // clamp: m (sx-idx) for A, n (sy-idx) for B

    v8f c = {0.f, 0.f, 0.f, 0.f, 0.f, 0.f, 0.f, 0.f};

    for (int rr = 0; rr < 4; ++rr) {
        const int hl = wv * 4 + rr;                       // local input row
        const __bf16* aRow = &sI[hl][0];                  // A value: aRow[w0+K+m] = I[h, w0+K+m-2]
        const __bf16* bRow = &sT[hl + 4 - idx5][0];       // B row: T[h - (n-2), *]

        for (int w0 = 0; w0 < WW; w0 += 32) {
            v16bf A, Bm;
#pragma unroll
            for (int v = 0; v < 8; ++v) {
#pragma unroll
                for (int u = 0; u < 2; ++u) {
                    // A 16x32 bf16: K = half*16 + v*2 + u (contiguous per lane, matches B)
                    int Ka = half * 16 + v * 2 + u;
                    A[v * 2 + u] = aRow[w0 + Ka + idx5];
                    // B 32x16 bf16: lane=N, K = half*16 + v*2 + u (contiguous per lane)
                    int Kb = half * 16 + v * 2 + u;
                    Bm[v * 2 + u] = bRow[w0 + Kb];
                }
            }
            c = __builtin_amdgcn_wmma_f32_16x16x32_bf16(
                    false, A, false, Bm, (short)0, c, false, false);
        }
    }

    // C 16x16 f32: VGPR v -> M=v (lanes 0-15), N=lane. M=sx-idx, N=sy-idx.
    if (lane < 5) {
#pragma unroll
        for (int v = 0; v < 5; ++v)
            atomicAdd(&sInter[v * 5 + lane], c[v]);
    }
    __syncthreads();

    // ---- Write per-WG partials: ws[j*NWG + wg], j<25 intersections, 25/26 sums ----
    if (tid < 27) {
        const int wg = img * NBLK + blk;
        float val = (tid < 25) ? sInter[tid] : sSum[tid - 25];
        wsf[(size_t)tid * NWG + wg] = val;
    }
}

__global__ __launch_bounds__(256)
void dice_finalize(const float* __restrict__ wsf, float* __restrict__ out) {
    __shared__ float tot[27];
    const int tid = threadIdx.x;
    if (tid < 27) tot[tid] = 0.f;
    __syncthreads();

    for (int j = 0; j < 27; ++j) {
        float s = 0.f;
        for (int i = tid; i < NWG; i += 256) s += wsf[(size_t)j * NWG + i];
        atomicAdd(&tot[j], s);
    }
    __syncthreads();

    if (tid == 0) {
        float denom = tot[25] + tot[26] + 1.0f;   // i_sum + t_sum + SMOOTH
        float best = tot[0];
        for (int j = 1; j < 25; ++j) best = fmaxf(best, tot[j]);
        out[0] = 1.0f - (2.0f * best + 1.0f) / denom;   // min loss = max intersection
    }
}

extern "C" void kernel_launch(void* const* d_in, const int* in_sizes, int n_in,
                              void* d_out, int out_size, void* d_ws, size_t ws_size,
                              hipStream_t stream) {
    const float* inp = (const float*)d_in[0];
    const float* tgt = (const float*)d_in[1];
    float* wsf = (float*)d_ws;     // 27 * NWG floats = 108 KB
    float* outp = (float*)d_out;

    dim3 grid(NBLK, NIMG);
    dice_main<<<grid, 256, 0, stream>>>(inp, tgt, wsf);
    dice_finalize<<<1, 256, 0, stream>>>(wsf, outp);
}
